// CrossAttention_73890617360795
// MI455X (gfx1250) — compile-verified, matrix operand order checked by reading, not run on hardware
//
#include <hip/hip_runtime.h>
#include <hip/hip_bf16.h>

// CDNA5 / gfx1250, wave32. Matrix math via v_wmma_f32_16x16x32_f16; K/V tiles
// staged global->LDS with async copies (ASYNCcnt) double-buffered per wave.
typedef _Float16 half_t;
typedef __attribute__((ext_vector_type(16))) _Float16 v16h;
typedef __attribute__((ext_vector_type(8)))  _Float16 v8h;
typedef __attribute__((ext_vector_type(8)))  float    v8f;

#define WMMA_F16(a, b, c) \
  __builtin_amdgcn_wmma_f32_16x16x32_f16(false, (a), false, (b), (short)0, (c), false, false)

// ---------------------------------------------------------------------------
// Fragment loader. For both A-fragments (lane holds row M=lane&15) and
// B-fragments (lane holds col N=lane&15), the per-lane K pattern is:
//   e=0..7  -> K = hb*8 + e          (hb = lane>=16)
//   e=8..15 -> K = 16 + hb*8 + (e-8)
// i.e. two contiguous 16-byte chunks at +hb*8 and +16+hb*8 halfs from the
// lane's row pointer. Works for global (global_load_b128) and LDS
// (ds_load_b128) source pointers.
// ---------------------------------------------------------------------------
static __device__ __forceinline__ v16h load_frag_f16(const half_t* p) {
  const int hb = (threadIdx.x >> 4) & 1;
  v8h c0 = *(const v8h*)(p + hb * 8);
  v8h c1 = *(const v8h*)(p + 16 + hb * 8);
  v16h r;
#pragma unroll
  for (int i = 0; i < 8; ++i) { r[i] = c0[i]; r[i + 8] = c1[i]; }
  return r;
}

// Same pattern, fp32 source, converted to f16 in-register.
static __device__ __forceinline__ v16h load_frag_f32(const float* p) {
  const int hb = (threadIdx.x >> 4) & 1;
  const float* q0 = p + hb * 8;
  const float* q1 = p + 16 + hb * 8;
  v16h r;
#pragma unroll
  for (int i = 0; i < 8; ++i) { r[i] = (half_t)q0[i]; r[i + 8] = (half_t)q1[i]; }
  return r;
}

// Async global->LDS copy of 16 bytes per lane (tracked by ASYNCcnt).
// ldsb = LDS byte address (low 32 bits of a generic pointer into __shared__
// are the LDS offset per the aperture mapping); g = per-lane global pointer.
static __device__ __forceinline__ void async_cp_b128(unsigned ldsb, const half_t* g) {
  asm volatile("global_load_async_to_lds_b128 %0, %1, off"
               :: "v"(ldsb), "v"(g) : "memory");
}

// ---------------------------------------------------------------------------
// Weight transpose + f32->f16 convert: Wt[n][k] = (f16) W[k][n].
// ---------------------------------------------------------------------------
__global__ __launch_bounds__(256) void transpose_cvt_kernel(
    const float* __restrict__ W, half_t* __restrict__ Wt, int K, int N) {
  __shared__ half_t tile[32][33];
  const int n0 = blockIdx.x * 32;
  const int k0 = blockIdx.y * 32;
#pragma unroll
  for (int i = threadIdx.y; i < 32; i += 8)
    tile[i][threadIdx.x] = (half_t)W[(size_t)(k0 + i) * N + (n0 + threadIdx.x)];
  __syncthreads();
#pragma unroll
  for (int i = threadIdx.y; i < 32; i += 8)
    Wt[(size_t)(n0 + i) * K + (k0 + threadIdx.x)] = tile[threadIdx.x][i];
}

// ---------------------------------------------------------------------------
// WMMA GEMM: C[M,N] = A[M,K] * W[K,N] + bias,  W given transposed (Wt[n][k]).
// Block = 256 threads = 8 waves; block tile 128(M) x 64(N); wave tile 16x64.
// A_IS_F16: A f16 row-major, else fp32 row-major (converted in-register).
// OUT_MODE: 0 = f16 row-major; 1 = f16 transposed per batch (C[b][n][s]);
//           2 = f32 row-major.
// ---------------------------------------------------------------------------
template <bool A_IS_F16, int OUT_MODE>
__global__ __launch_bounds__(256) void gemm_wmma_kernel(
    const void* __restrict__ Ap, const half_t* __restrict__ Wt,
    const float* __restrict__ bias, void* __restrict__ Cp,
    int M, int N, int K, int batch_rows) {
  const int wave = threadIdx.x >> 5;
  const int lane = threadIdx.x & 31;
  const int ln = lane & 15;
  const int hb = (lane >> 4) & 1;
  const int m0 = blockIdx.x * 128 + wave * 16;
  const int n0 = blockIdx.y * 64;

  v8f acc[4] = {};
  const half_t* A16 = (const half_t*)Ap;
  const float*  A32 = (const float*)Ap;
  const size_t arow = (size_t)(m0 + ln) * K;

  for (int k0 = 0; k0 < K; k0 += 32) {
    v16h af;
    if (A_IS_F16) {
      af = load_frag_f16(A16 + arow + k0);
      if (k0 + 256 < K) __builtin_prefetch(A16 + arow + k0 + 256, 0, 3);
    } else {
      af = load_frag_f32(A32 + arow + k0);
      if (k0 + 256 < K) __builtin_prefetch(A32 + arow + k0 + 256, 0, 3);
    }
#pragma unroll
    for (int t = 0; t < 4; ++t) {
      v16h bf = load_frag_f16(Wt + (size_t)(n0 + t * 16 + ln) * K + k0);
      acc[t] = WMMA_F16(af, bf, acc[t]);
    }
  }

#pragma unroll
  for (int t = 0; t < 4; ++t) {
    const int n = n0 + t * 16 + ln;
    const float bv = bias[n];
    if (OUT_MODE == 0) {               // f16 row-major (Q, K)
      half_t* C = (half_t*)Cp;
#pragma unroll
      for (int r = 0; r < 8; ++r)
        C[(size_t)(m0 + hb * 8 + r) * N + n] = (half_t)(acc[t][r] + bv);
    } else if (OUT_MODE == 1) {        // f16 transposed per batch (V)
      half_t* C = (half_t*)Cp;
      const int bidx = m0 / batch_rows;
      const int s0 = m0 - bidx * batch_rows + hb * 8;
      v8h o;
#pragma unroll
      for (int r = 0; r < 8; ++r) o[r] = (half_t)(acc[t][r] + bv);
      *(v8h*)(C + ((size_t)bidx * N + n) * batch_rows + s0) = o;
    } else {                           // f32 row-major (final output)
      float* C = (float*)Cp;
#pragma unroll
      for (int r = 0; r < 8; ++r)
        C[(size_t)(m0 + hb * 8 + r) * N + n] = acc[t][r] + bv;
    }
  }
}

// ---------------------------------------------------------------------------
// Flash attention, transposed formulation (S^T = K*Q^T so softmax is in-lane
// + one xor-16 shuffle, and exp'd P^T in C-layout IS the B-fragment for
// O^T += V^T * P^T). K/V kv-tiles are double-buffered in LDS via
// global_load_async_to_lds_b128 (ASYNCcnt): issue tile i+1, wait <=16 (its
// own 16 issues; in-order completion ==> tile i resident), compute tile i.
// 4 waves/block; per-wave slice: 2 x (4KB K[32x64] + 4KB Vt[64x32]) = 16KB.
// ---------------------------------------------------------------------------
__global__ __launch_bounds__(128) void flash_attn_kernel(
    const half_t* __restrict__ Q, const half_t* __restrict__ Kh,
    const half_t* __restrict__ Vt, half_t* __restrict__ O) {
  __shared__ __align__(16) half_t smem[4][2][4096];  // [wave][buf][K 2048 | V 2048]

  const int wave = threadIdx.x >> 5;
  const int lane = threadIdx.x & 31;
  const int ln = lane & 15;
  const int hb = (lane >> 4) & 1;
  const int b = blockIdx.z, h = blockIdx.y;
  const int q0 = blockIdx.x * 64 + wave * 16;
  const int Hn = 1024, HD = 64, SKV = 2048, SQn = 2048;

  const half_t* Kbase = Kh + (size_t)b * SKV * Hn + h * HD;
  const half_t* Vbase = Vt + ((size_t)b * Hn + h * HD) * SKV;
  const unsigned lds0 = (unsigned)(uintptr_t)&smem[wave][0][0];
  const unsigned lds1 = (unsigned)(uintptr_t)&smem[wave][1][0];

  // Stage one kv tile (32 rows K, 64 rows V^T): 8+8 async b128 issues.
  auto stage = [&](unsigned base, int kv0) {
    // K: 32 rows x 128B; lane = row, 8 chunks across the row.
    const half_t* kg = Kbase + (size_t)(kv0 + lane) * Hn;
    const unsigned kd = base + (unsigned)lane * 128u;
#pragma unroll
    for (int i = 0; i < 8; ++i) async_cp_b128(kd + i * 16u, kg + i * 8);
    // V^T: 64 rows x 64B; 4 lanes per row, 8 row-groups.
    const half_t* vg = Vbase + (size_t)(lane >> 2) * SKV + kv0 + (lane & 3) * 8;
    const unsigned vd = base + 4096u + (unsigned)(lane >> 2) * 64u + (unsigned)(lane & 3) * 16u;
#pragma unroll
    for (int i = 0; i < 8; ++i) async_cp_b128(vd + i * 8u * 64u, vg + (size_t)i * 8 * SKV);
  };

  // Q^T B-fragments (lane = q column), held for the whole kv loop.
  const half_t* qrow = Q + (size_t)(b * SQn + q0 + ln) * Hn + h * HD;
  const v16h qf0 = load_frag_f16(qrow);
  const v16h qf1 = load_frag_f16(qrow + 32);

  float m = -3.0e38f, l = 0.0f;
  v8f acc[4] = {};
  const float sc  = 0.125f;        // 1/sqrt(64)
  const float l2e = 1.44269504088896340736f;

  stage(lds0, 0);  // prime buffer 0

  for (int it = 0; it < SKV / 32; ++it) {
    const int buf = it & 1;
    const bool more = (it + 1) < (SKV / 32);
    if (more) {
      stage(buf ? lds0 : lds1, (it + 1) * 32);
      asm volatile("s_wait_asynccnt 0x10" ::: "memory");  // current tile resident
    } else {
      asm volatile("s_wait_asynccnt 0x0" ::: "memory");
    }
    const half_t* kt = &smem[wave][buf][0];     // K tile  [32][64] halfs
    const half_t* vt = kt + 2048;               // V^T tile [64][32] halfs

    // S^T tiles: s0 = kv rows 0..15, s1 = 16..31 of this tile (N = q).
    v8f s0 = {}, s1 = {};
    {
      v16h a00 = load_frag_f16(kt + (size_t)ln * 64);
      v16h a10 = load_frag_f16(kt + (size_t)(16 + ln) * 64);
      s0 = WMMA_F16(a00, qf0, s0);
      s1 = WMMA_F16(a10, qf0, s1);
      v16h a01 = load_frag_f16(kt + (size_t)ln * 64 + 32);
      v16h a11 = load_frag_f16(kt + (size_t)(16 + ln) * 64 + 32);
      s0 = WMMA_F16(a01, qf1, s0);
      s1 = WMMA_F16(a11, qf1, s1);
    }
    // Online softmax over kv: 16 in-lane values + partner half-wave (xor 16).
    float tmax = -3.0e38f;
#pragma unroll
    for (int r = 0; r < 8; ++r) tmax = fmaxf(tmax, fmaxf(s0[r], s1[r]));
    tmax = fmaxf(tmax, __shfl_xor(tmax, 16, 32));
    const float mnew = fmaxf(m, tmax * sc);
    const float fs = sc * l2e, fb = -mnew * l2e;
    float rsum = 0.0f;
    v8h p0, p1;
#pragma unroll
    for (int r = 0; r < 8; ++r) {
      float e0 = exp2f(s0[r] * fs + fb);
      float e1 = exp2f(s1[r] * fs + fb);
      rsum += e0 + e1;
      p0[r] = (half_t)e0;
      p1[r] = (half_t)e1;
    }
    rsum += __shfl_xor(rsum, 16, 32);
    const float alpha = exp2f((m - mnew) * l2e);
    l = l * alpha + rsum;
    // P^T C-layout -> B-fragment: pure in-lane repack (layouts coincide).
    v16h pb;
#pragma unroll
    for (int i = 0; i < 8; ++i) { pb[i] = p0[i]; pb[i + 8] = p1[i]; }
#pragma unroll
    for (int dt = 0; dt < 4; ++dt) {
#pragma unroll
      for (int r = 0; r < 8; ++r) acc[dt][r] *= alpha;
      v16h av = load_frag_f16(vt + (size_t)(dt * 16 + ln) * 32);  // V^T A-frag
      acc[dt] = WMMA_F16(av, pb, acc[dt]);
    }
    m = mnew;
  }

  // O^T C-layout: lane = q, rows = d -> contiguous packed f16 stores.
  const float inv = 1.0f / l;
  half_t* orow = O + (size_t)(b * SQn + q0 + ln) * Hn + h * HD;
#pragma unroll
  for (int dt = 0; dt < 4; ++dt) {
    v8h o;
#pragma unroll
    for (int r = 0; r < 8; ++r) o[r] = (half_t)(acc[dt][r] * inv);
    *(v8h*)(orow + dt * 16 + hb * 8) = o;
  }
}

// ---------------------------------------------------------------------------
// Host launcher. Workspace layout (72 MiB total):
//   [0,2M) WqT  [2M,4M) WkT  [4M,6M) WvT  [6M,8M) WoT   (f16, transposed)
//   [8M,24M) Q f16  [24M,40M) K f16  [40M,56M) V f16 transposed
//   [56M,72M) attention output f16
// ---------------------------------------------------------------------------
extern "C" void kernel_launch(void* const* d_in, const int* in_sizes, int n_in,
                              void* d_out, int out_size, void* d_ws, size_t ws_size,
                              hipStream_t stream) {
  const float* query     = (const float*)d_in[0];
  const float* key_value = (const float*)d_in[1];
  const float* Wq = (const float*)d_in[2];
  const float* bq = (const float*)d_in[3];
  const float* Wk = (const float*)d_in[4];
  const float* bk = (const float*)d_in[5];
  const float* Wv = (const float*)d_in[6];
  const float* bv = (const float*)d_in[7];
  const float* Wo = (const float*)d_in[8];
  const float* bo = (const float*)d_in[9];
  float* out = (float*)d_out;

  const int B = 4, SQ = 2048, SKV = 2048, H = 1024, D = 1024;
  const int M = B * SQ;                       // 8192
  char* ws = (char*)d_ws;
  const size_t WB = (size_t)D * H * sizeof(half_t);   // 2 MiB per weight
  const size_t AB = (size_t)M * H * sizeof(half_t);   // 16 MiB per activation
  half_t* WqT = (half_t*)(ws + 0 * WB);
  half_t* WkT = (half_t*)(ws + 1 * WB);
  half_t* WvT = (half_t*)(ws + 2 * WB);
  half_t* WoT = (half_t*)(ws + 3 * WB);
  half_t* Qh  = (half_t*)(ws + 4 * WB + 0 * AB);
  half_t* Kh  = (half_t*)(ws + 4 * WB + 1 * AB);
  half_t* Vth = (half_t*)(ws + 4 * WB + 2 * AB);
  half_t* Ao  = (half_t*)(ws + 4 * WB + 3 * AB);

  // 1) weights -> f16, transposed
  dim3 tb(32, 8), tg(H / 32, D / 32);
  transpose_cvt_kernel<<<tg, tb, 0, stream>>>(Wq, WqT, D, H);
  transpose_cvt_kernel<<<tg, tb, 0, stream>>>(Wk, WkT, D, H);
  transpose_cvt_kernel<<<tg, tb, 0, stream>>>(Wv, WvT, D, H);
  transpose_cvt_kernel<<<tg, tb, 0, stream>>>(Wo, WoT, H, D);

  // 2) projections (fp32 A, f16 out; V stored transposed per batch)
  dim3 gb(256), gg(M / 128, H / 64);
  gemm_wmma_kernel<false, 0><<<gg, gb, 0, stream>>>(query,     WqT, bq, Qh,  M, H, D, SQ);
  gemm_wmma_kernel<false, 0><<<gg, gb, 0, stream>>>(key_value, WkT, bk, Kh,  M, H, D, SKV);
  gemm_wmma_kernel<false, 1><<<gg, gb, 0, stream>>>(key_value, WvT, bv, Vth, M, H, D, SKV);

  // 3) flash attention: grid (SQ/64 q-tiles, NH, B), 4 waves/block
  flash_attn_kernel<<<dim3(SQ / 64, 16, B), 128, 0, stream>>>(Qh, Kh, Vth, Ao);

  // 4) output projection (f16 A, fp32 out + bias)
  gemm_wmma_kernel<true, 2><<<gg, gb, 0, stream>>>(Ao, WoT, bo, out, M, D, H, SQ);
}
